// WindowAttention_89687507075201
// MI455X (gfx1250) — matrix operand order repaired and checked by reading, not compiled, not run on hardware
//
#include <hip/hip_runtime.h>
#include <hip/hip_bf16.h>
#include <math.h>

// ---------------- problem constants ----------------
#define DIMC   384
#define HEADS  12
#define NTOK   144          // tokens per window (12*12)
#define HD     32           // head dim
#define BATCH  512          // B_
#define NWIN   64
#define ROWS   (BATCH*NTOK) // 73728
#define NQKV   (3*DIMC)     // 1152
#define CPB_PTS 529         // (2*12-1)^2
#define CPB_HID 512
#define NPAD   160          // 144 padded to mult of 32 for P@V K-loop

typedef __attribute__((ext_vector_type(16))) __bf16 v16bf;
typedef __attribute__((ext_vector_type(8)))  __bf16 v8bf;
typedef __attribute__((ext_vector_type(8)))  float  v8f;

// ---------------- WMMA / async helpers (gfx1250, wave32) ----------------
__device__ __forceinline__ v8f wmma_bf16(v16bf a, v16bf b, v8f c) {
  // D = A(16x32) * B(32x16) + C(16x16 f32)
  return __builtin_amdgcn_wmma_f32_16x16x32_bf16(false, a, false, b, (short)0, c,
                                                 false, false);
}

// low 32 bits of a flat LDS pointer == wave-relative LDS byte offset (ISA 10.2)
__device__ __forceinline__ unsigned lds_off_u32(const void* p) {
  return (unsigned)(uintptr_t)p;
}
// 16B global -> LDS async copy, tracked by ASYNCcnt (CDNA5 ISA)
__device__ __forceinline__ void async_ld128(unsigned ldsoff, const void* g) {
  asm volatile("global_load_async_to_lds_b128 %0, %1, off"
               :: "v"(ldsoff), "v"(g) : "memory");
}
__device__ __forceinline__ void wait_async0() {
  asm volatile("s_wait_asynccnt 0x0" ::: "memory");
}

// A-matrix 16x32 bf16 fragment. rowk points at this lane's row start + k-tile base.
// Layout: lanes 0-15 hold K 0..15, lanes 16-31 hold K 16..31 (contiguous, matches B).
__device__ __forceinline__ v16bf load_a_frag(const __bf16* rowk) {
  const int lane = threadIdx.x & 31;
  const int off  = (lane & 16) ? 16 : 0;
  v8bf lo = *(const v8bf*)(rowk + off);
  v8bf hi = *(const v8bf*)(rowk + off + 8);
  v16bf r;
#pragma unroll
  for (int i = 0; i < 8; ++i) { r[i] = lo[i]; r[i + 8] = hi[i]; }
  return r;
}

// B-matrix 32x16 bf16 fragment. colk points at this lane's column data (K-major,
// contiguous K). Lanes 0-15: K 0..15 ; lanes 16-31: K 16..31.
__device__ __forceinline__ v16bf load_b_frag(const __bf16* colk) {
  const int lane = threadIdx.x & 31;
  const int off  = (lane & 16) ? 16 : 0;
  v8bf lo = *(const v8bf*)(colk + off);
  v8bf hi = *(const v8bf*)(colk + off + 8);
  v16bf r;
#pragma unroll
  for (int i = 0; i < 8; ++i) { r[i] = lo[i]; r[i + 8] = hi[i]; }
  return r;
}

// ---------------- kernel 1: repack weight matrix into WMMA-ready B layout -------
// pack[((kt*ntiles + nt)*32 + lane)*16 + i]  = w[K, N]  with
//   K = kt*32 + (lane<16 ? i : 16+i),  N = nt*16 + (lane&15)
__global__ void k_repack_b(const float* __restrict__ w, __bf16* __restrict__ pack,
                           int K, int Ncols) {
  const int ntiles = Ncols >> 4;
  const int total  = K * Ncols;
  for (int p = blockIdx.x * blockDim.x + threadIdx.x; p < total;
       p += gridDim.x * blockDim.x) {
    int i    = p & 15;
    int lane = (p >> 4) & 31;
    int nt   = (p >> 9) % ntiles;
    int kt   = (p >> 9) / ntiles;
    int Kidx = kt * 32 + ((lane & 16) ? 16 + i : i);
    int Nidx = nt * 16 + (lane & 15);
    pack[p] = (__bf16)w[Kidx * Ncols + Nidx];
  }
}

// ---------------- kernel 2: CPB MLP table t[529][12] ----------------
__global__ void k_cpb_mlp(const float* __restrict__ tbl, const float* __restrict__ w1,
                          const float* __restrict__ b1, const float* __restrict__ w2,
                          float* __restrict__ t) {
  int j = blockIdx.x * blockDim.x + threadIdx.x;
  if (j >= CPB_PTS) return;
  float c0 = tbl[j * 2 + 0], c1 = tbl[j * 2 + 1];
  float acc[HEADS];
#pragma unroll
  for (int h = 0; h < HEADS; ++h) acc[h] = 0.f;
  for (int u = 0; u < CPB_HID; ++u) {
    float hid = fmaxf(0.f, c0 * w1[u] + c1 * w1[CPB_HID + u] + b1[u]);
#pragma unroll
    for (int h = 0; h < HEADS; ++h) acc[h] += hid * w2[u * HEADS + h];
  }
#pragma unroll
  for (int h = 0; h < HEADS; ++h) t[j * HEADS + h] = acc[h];
}

// ---------------- kernel 3: gather + 16*sigmoid -> bias[H][144][144] ------------
__global__ void k_cpb_bias(const float* __restrict__ t, const int* __restrict__ idx,
                           float* __restrict__ bias) {
  int p = blockIdx.x * blockDim.x + threadIdx.x;
  if (p >= HEADS * NTOK * NTOK) return;
  int h  = p / (NTOK * NTOK);
  int rc = p % (NTOK * NTOK);
  float v = t[idx[rc] * HEADS + h];
  bias[p] = 16.f / (1.f + expf(-v));
}

// ---------------- epilogue helpers ----------------
__device__ __forceinline__ void qkv_store(v8f acc, int nt, int lane, int row0,
                                          const float* __restrict__ qkvb,
                                          __bf16* __restrict__ qb,
                                          __bf16* __restrict__ kb,
                                          __bf16* __restrict__ vb) {
  const int ncol  = nt * 16 + (lane & 15);
  const float bv  = qkvb[ncol];
  const int which = ncol / DIMC, cc = ncol % DIMC, h = cc >> 5, d = cc & 31;
  __bf16* dst = (which == 0) ? qb : (which == 1) ? kb : vb;
#pragma unroll
  for (int r = 0; r < 8; ++r) {
    int grow = row0 + r + ((lane & 16) ? 8 : 0);
    int bidx = grow / NTOK, n = grow % NTOK;
    size_t o = ((size_t)(bidx * HEADS + h) * NTOK + n) * HD + d;
    dst[o] = (__bf16)(acc[r] + bv);
  }
}

__device__ __forceinline__ void score_store(v8f c, int mt, int nt, int lane, float ls,
                                            const float* __restrict__ biash,
                                            const float* __restrict__ maskw,
                                            float* S) {
  const int n = nt * 16 + (lane & 15);
#pragma unroll
  for (int rr = 0; rr < 8; ++rr) {
    int m = mt * 16 + rr + ((lane & 16) ? 8 : 0);
    S[m * NTOK + n] = c[rr] * ls + biash[m * NTOK + n] + maskw[m * NTOK + n];
  }
}

__device__ __forceinline__ void proj_store(v8f acc, int nt, int lane, int row0,
                                           const float* __restrict__ pb,
                                           float* __restrict__ out) {
  const int col = nt * 16 + (lane & 15);
  const float bv = pb[col];
#pragma unroll
  for (int rr = 0; rr < 8; ++rr) {
    int m = rr + ((lane & 16) ? 8 : 0);
    out[(size_t)(row0 + m) * DIMC + col] = acc[rr] + bv;
  }
}

// ---------------- kernel 4: QKV GEMM 73728x384 @ 384x1152 ----------------
// one 16-row tile per workgroup; 8 waves x (3 groups x 3 N-tiles) x 12 K-step WMMA
__global__ __launch_bounds__(256) void k_qkv_gemm(
    const float* __restrict__ x, const __bf16* __restrict__ packw,
    const float* __restrict__ qkvb, __bf16* __restrict__ qb,
    __bf16* __restrict__ kb, __bf16* __restrict__ vb) {
  __shared__ __align__(16) float  xs[16 * DIMC];   // 24576 B, async-staged
  __shared__ __align__(16) __bf16 As[16 * DIMC];   // 12288 B
  const int row0 = blockIdx.x * 16;
  const float* gx = x + (size_t)row0 * DIMC;
  const unsigned lxs = lds_off_u32(xs);
  for (int c = threadIdx.x; c < (16 * DIMC) / 4; c += 256)   // 1536 x 16B
    async_ld128(lxs + c * 16, gx + c * 4);
  wait_async0();
  __syncthreads();
  for (int e = threadIdx.x; e < 16 * DIMC; e += 256)
    As[e] = (__bf16)xs[e];
  __syncthreads();

  const int wave = threadIdx.x >> 5, lane = threadIdx.x & 31;
  const int mrow = lane & 15;
#pragma unroll 1
  for (int g = 0; g < 3; ++g) {
    const int nt = wave * 9 + g * 3;
    const __bf16* pw0 = packw + (size_t)((nt + 0) * 32 + lane) * 16;
    const __bf16* pw1 = packw + (size_t)((nt + 1) * 32 + lane) * 16;
    const __bf16* pw2 = packw + (size_t)((nt + 2) * 32 + lane) * 16;
    v8f c0 = {}, c1 = {}, c2 = {};
#pragma unroll
    for (int kt = 0; kt < 12; ++kt) {
      v16bf a  = load_a_frag(&As[mrow * DIMC + kt * 32]);
      v16bf b0 = *(const v16bf*)(pw0 + (size_t)kt * (72 * 512));
      v16bf b1 = *(const v16bf*)(pw1 + (size_t)kt * (72 * 512));
      v16bf b2 = *(const v16bf*)(pw2 + (size_t)kt * (72 * 512));
      c0 = wmma_bf16(a, b0, c0);
      c1 = wmma_bf16(a, b1, c1);
      c2 = wmma_bf16(a, b2, c2);
    }
    qkv_store(c0, nt + 0, lane, row0, qkvb, qb, kb, vb);
    qkv_store(c1, nt + 1, lane, row0, qkvb, qb, kb, vb);
    qkv_store(c2, nt + 2, lane, row0, qkvb, qb, kb, vb);
  }
}

// ---------------- kernel 5: windowed cosine attention (one WG per (b,h)) --------
__global__ __launch_bounds__(160) void k_attention(
    const __bf16* __restrict__ qb, const __bf16* __restrict__ kb,
    const __bf16* __restrict__ vb, const float* __restrict__ bias,
    const float* __restrict__ mask, const float* __restrict__ logit_scale,
    __bf16* __restrict__ aout) {
  __shared__ __align__(16) __bf16 qs[NTOK * HD];   //  9216 B
  __shared__ __align__(16) __bf16 ks[NTOK * HD];   //  9216 B
  __shared__ __align__(16) __bf16 vT[HD * NPAD];   // 10240 B (V^T, K-padded)
  __shared__ __align__(16) float  S[NTOK * NTOK];  // 82944 B
  __shared__ __align__(16) __bf16 P[NTOK * NPAD];  // 46080 B
  __bf16* vs = P;  // V row-major staging aliases P (P written only after vs is dead)

  const int b = blockIdx.x / HEADS, h = blockIdx.x % HEADS;
  const size_t base = (size_t)(b * HEADS + h) * NTOK * HD;
  const __bf16* gq = qb + base;
  const __bf16* gk = kb + base;
  const __bf16* gv = vb + base;
  const unsigned lq = lds_off_u32(qs), lk = lds_off_u32(ks), lv = lds_off_u32(vs);
  for (int c = threadIdx.x; c < (NTOK * HD) / 8; c += 160) {  // 576 x 16B each
    async_ld128(lq + c * 16, gq + c * 8);
    async_ld128(lk + c * 16, gk + c * 8);
    async_ld128(lv + c * 16, gv + c * 8);
  }
  wait_async0();
  __syncthreads();

  // transpose V into vT (+ zero K-pad rows); cosine-normalize q,k rows in LDS
  for (int e = threadIdx.x; e < NTOK * HD; e += 160) {
    int m = e >> 5, d = e & 31;
    vT[d * NPAD + m] = vs[e];
  }
  for (int e = threadIdx.x; e < HD * (NPAD - NTOK); e += 160)
    vT[(e / (NPAD - NTOK)) * NPAD + NTOK + (e % (NPAD - NTOK))] = (__bf16)0.f;
  const int r = threadIdx.x;
  if (r < NTOK) {
    float sq = 0.f, sk = 0.f;
#pragma unroll
    for (int d = 0; d < HD; ++d) {
      float a = (float)qs[r * HD + d]; sq += a * a;
      float c = (float)ks[r * HD + d]; sk += c * c;
    }
    float iq = rsqrtf(fmaxf(sq, 1e-24f)), ik = rsqrtf(fmaxf(sk, 1e-24f));
#pragma unroll
    for (int d = 0; d < HD; ++d) {
      qs[r * HD + d] = (__bf16)((float)qs[r * HD + d] * iq);
      ks[r * HD + d] = (__bf16)((float)ks[r * HD + d] * ik);
    }
  }
  const float ls = expf(fminf(fmaxf(logit_scale[h], -100.f), logf(100.f)));
  const int wi = b % NWIN;
  const float* biash = bias + h * NTOK * NTOK;
  const float* maskw = mask + wi * NTOK * NTOK;
  __syncthreads();

  const int wave = threadIdx.x >> 5, lane = threadIdx.x & 31;
  // scores: per M-tile, A fragment shared across 9 N-tiles; 3 accumulators in flight
  for (int mt = wave; mt < 9; mt += 5) {
    v16bf a = load_a_frag(&qs[(mt * 16 + (lane & 15)) * HD]);
#pragma unroll
    for (int n3 = 0; n3 < 9; n3 += 3) {
      v16bf b0 = load_b_frag(&ks[((n3 + 0) * 16 + (lane & 15)) * HD]);
      v16bf b1 = load_b_frag(&ks[((n3 + 1) * 16 + (lane & 15)) * HD]);
      v16bf b2 = load_b_frag(&ks[((n3 + 2) * 16 + (lane & 15)) * HD]);
      v8f c0 = {}, c1 = {}, c2 = {};
      c0 = wmma_bf16(a, b0, c0);
      c1 = wmma_bf16(a, b1, c1);
      c2 = wmma_bf16(a, b2, c2);
      score_store(c0, mt, n3 + 0, lane, ls, biash, maskw, S);
      score_store(c1, mt, n3 + 1, lane, ls, biash, maskw, S);
      score_store(c2, mt, n3 + 2, lane, ls, biash, maskw, S);
    }
  }
  __syncthreads();

  // row softmax (f32) -> P (bf16, K-padded with zeros)
  if (r < NTOK) {
    float mx = -3.0e38f;
    for (int c = 0; c < NTOK; ++c) mx = fmaxf(mx, S[r * NTOK + c]);
    float sum = 0.f;
    for (int c = 0; c < NTOK; ++c) {
      float e = expf(S[r * NTOK + c] - mx);
      S[r * NTOK + c] = e;
      sum += e;
    }
    float inv = 1.f / sum;
    for (int c = 0; c < NTOK; ++c) P[r * NPAD + c] = (__bf16)(S[r * NTOK + c] * inv);
#pragma unroll
    for (int c = NTOK; c < NPAD; ++c) P[r * NPAD + c] = (__bf16)0.f;
  }
  __syncthreads();

  // O = P(144x160) @ V(160x32): per M-tile both D-tiles share the A fragment
  for (int mt = wave; mt < 9; mt += 5) {
    v8f o0 = {}, o1 = {};
#pragma unroll
    for (int kt = 0; kt < 5; ++kt) {
      v16bf a  = load_a_frag(&P[(mt * 16 + (lane & 15)) * NPAD + kt * 32]);
      v16bf b0 = load_b_frag(&vT[((lane & 15)) * NPAD + kt * 32]);
      v16bf b1 = load_b_frag(&vT[((16 + (lane & 15))) * NPAD + kt * 32]);
      o0 = wmma_bf16(a, b0, o0);
      o1 = wmma_bf16(a, b1, o1);
    }
#pragma unroll
    for (int rr = 0; rr < 8; ++rr) {
      int m = mt * 16 + rr + ((lane & 16) ? 8 : 0);
      size_t row = (size_t)b * NTOK + m;
      aout[row * DIMC + h * HD + (lane & 15)]      = (__bf16)o0[rr];
      aout[row * DIMC + h * HD + 16 + (lane & 15)] = (__bf16)o1[rr];
    }
  }
}

// ---------------- kernel 6: proj GEMM 73728x384 @ 384x384 + bias ----------------
__global__ __launch_bounds__(256) void k_proj_gemm(
    const __bf16* __restrict__ aout, const __bf16* __restrict__ packw,
    const float* __restrict__ pb, float* __restrict__ out) {
  const int row0 = blockIdx.x * 16;
  const int wave = threadIdx.x >> 5, lane = threadIdx.x & 31;
  const int nt = wave * 3;
  const __bf16* pw0 = packw + (size_t)((nt + 0) * 32 + lane) * 16;
  const __bf16* pw1 = packw + (size_t)((nt + 1) * 32 + lane) * 16;
  const __bf16* pw2 = packw + (size_t)((nt + 2) * 32 + lane) * 16;
  const __bf16* arow = aout + (size_t)(row0 + (lane & 15)) * DIMC;
  v8f c0 = {}, c1 = {}, c2 = {};
#pragma unroll
  for (int kt = 0; kt < 12; ++kt) {
    v16bf a  = load_a_frag(arow + kt * 32);
    v16bf b0 = *(const v16bf*)(pw0 + (size_t)kt * (24 * 512));
    v16bf b1 = *(const v16bf*)(pw1 + (size_t)kt * (24 * 512));
    v16bf b2 = *(const v16bf*)(pw2 + (size_t)kt * (24 * 512));
    c0 = wmma_bf16(a, b0, c0);
    c1 = wmma_bf16(a, b1, c1);
    c2 = wmma_bf16(a, b2, c2);
  }
  proj_store(c0, nt + 0, lane, row0, pb, out);
  proj_store(c1, nt + 1, lane, row0, pb, out);
  proj_store(c2, nt + 2, lane, row0, pb, out);
}

// ---------------- launch ----------------
extern "C" void kernel_launch(void* const* d_in, const int* in_sizes, int n_in,
                              void* d_out, int out_size, void* d_ws, size_t ws_size,
                              hipStream_t stream) {
  const float* x     = (const float*)d_in[0];
  const float* mask  = (const float*)d_in[1];
  const float* qkv_w = (const float*)d_in[2];
  const float* qkv_b = (const float*)d_in[3];
  const float* pr_w  = (const float*)d_in[4];
  const float* pr_b  = (const float*)d_in[5];
  const float* cw1   = (const float*)d_in[6];
  const float* cb1   = (const float*)d_in[7];
  const float* cw2   = (const float*)d_in[8];
  const float* lsc   = (const float*)d_in[9];
  const float* rtbl  = (const float*)d_in[10];
  const int*   ridx  = (const int*)d_in[11];
  float* out = (float*)d_out;

  char* ws = (char*)d_ws;
  // 256B-aligned workspace layout
  const size_t OFF_PACKQKV  = 0;                   // 384*1152*2  = 884736
  const size_t OFF_PACKPROJ = 884736;              // 384*384*2   = 294912
  const size_t OFF_T        = 1179648;             // 529*12*4    ->  25600
  const size_t OFF_BIAS     = 1205248;             // 12*144*144*4=995328
  const size_t OFF_Q        = 2200576;             // 56623104
  const size_t OFF_K        = OFF_Q + 56623104;
  const size_t OFF_V        = OFF_K + 56623104;
  const size_t OFF_AOUT     = OFF_V + 56623104;

  __bf16* packq = (__bf16*)(ws + OFF_PACKQKV);
  __bf16* packp = (__bf16*)(ws + OFF_PACKPROJ);
  float*  tbuf  = (float*)(ws + OFF_T);
  float*  bias  = (float*)(ws + OFF_BIAS);
  __bf16* qbuf  = (__bf16*)(ws + OFF_Q);
  __bf16* kbuf  = (__bf16*)(ws + OFF_K);
  __bf16* vbuf  = (__bf16*)(ws + OFF_V);
  __bf16* abuf  = (__bf16*)(ws + OFF_AOUT);

  k_repack_b<<<1728, 256, 0, stream>>>(qkv_w, packq, DIMC, NQKV);
  k_repack_b<<<576, 256, 0, stream>>>(pr_w, packp, DIMC, DIMC);
  k_cpb_mlp<<<3, 256, 0, stream>>>(rtbl, cw1, cb1, cw2, tbuf);
  k_cpb_bias<<<(HEADS * NTOK * NTOK + 255) / 256, 256, 0, stream>>>(tbuf, ridx, bias);
  k_qkv_gemm<<<ROWS / 16, 256, 0, stream>>>(x, packq, qkv_b, qbuf, kbuf, vbuf);
  k_attention<<<BATCH * HEADS, 160, 0, stream>>>(qbuf, kbuf, vbuf, bias, mask, lsc, abuf);
  k_proj_gemm<<<ROWS / 16, 256, 0, stream>>>(abuf, packp, pr_b, out);
}
